// MultiScalePeriodDetector_15985868276406
// MI455X (gfx1250) — matrix operand order repaired and checked by reading, not compiled, not run on hardware
//
#include <hip/hip_runtime.h>
#include <hip/hip_bf16.h>
#include <math.h>

// ---------------- types for CDNA5 WMMA (wave32) ----------------
typedef __bf16 bf16_t;
typedef __attribute__((ext_vector_type(16))) __bf16 v16bf;
typedef __attribute__((ext_vector_type(8)))  float  v8f;
typedef __attribute__((ext_vector_type(4)))  int    v4i;
typedef __attribute__((ext_vector_type(8)))  int    v8i;

// ---------------- problem constants ----------------
#define BN     16            // batch
#define TT     512           // time length
#define CC     64            // channels
#define NROWS  1024          // B*C rows (series)
#define KTOP   5
#define TC     259           // db4 level-1 cA length = (512+7)/2
#define KW     272           // wavelet K dim padded to mult of 16
#define NW     160           // wavelet freq cols (>=130, mult of 32)
#define FW     130           // wavelet rfft bins = TC//2+1
#define FMAIN  257           // rfft bins for T=512
#define VK     3
#define VITERS 50

// reversed db4 decomposition low-pass filter (correlation form)
__constant__ float FREV[8] = {
     0.23037781330885523f,  0.7148465705525415f,  0.6308807679295904f,
    -0.02798376941698385f, -0.18703481171888114f, 0.030841381835986965f,
     0.032883011666982945f, -0.010597401784997278f };

// ---------------- small device helpers ----------------
__device__ inline float hash01(unsigned v) {
    v ^= v >> 17; v *= 0xed5ad4bbu; v ^= v >> 11; v *= 0xac4c1b51u;
    v ^= v >> 15; v *= 0x31848babu; v ^= v >> 14;
    return (float)(v >> 8) * (1.0f / 16777216.0f);
}

__device__ inline float wave_sum(float v) {
#pragma unroll
    for (int off = 16; off > 0; off >>= 1) v += __shfl_down(v, off, 32);
    return v;
}

__device__ inline void wave_argmax(float& v, int& i) {
#pragma unroll
    for (int off = 16; off > 0; off >>= 1) {
        float ov = __shfl_down(v, off, 32);
        int   oi = __shfl_down(i, off, 32);
        if (ov > v || (ov == v && oi < i)) { v = ov; i = oi; }
    }
}

// LDS byte offset of a __shared__ object: flat LDS aperture keeps the
// workgroup-relative offset in addr[31:0] (ISA 10.2 aperture table).
__device__ inline unsigned lds_offset_of(const void* p) {
    return (unsigned)(unsigned long long)p;
}

// ---------------- TDM: 2D tile (2-byte elements) -> LDS ----------------
// D# group0 (4 SGPRs) + group1 (8 SGPRs), 2-group form of tensor_load_to_lds
// (VADDR2/3 = NULL, tensors up to 2D). Inline asm instead of the clang builtin
// because the builtin arity differs across toolchains. Tracked by TENSORcnt.
__device__ inline void tdm_load_2d_b16(unsigned lds_byte_off, const void* gptr,
                                       unsigned tensor_d0, unsigned tensor_d1,
                                       unsigned tile_d0, unsigned tile_d1,
                                       unsigned row_stride_elems) {
    unsigned long long ga = (unsigned long long)gptr;
    v4i g0; v8i g1;
    g0[0] = 1;                                         // count=1, user descriptor
    g0[1] = (int)lds_byte_off;                         // lds_addr (bytes)
    g0[2] = (int)(unsigned)(ga & 0xffffffffull);       // global_addr[31:0]
    g0[3] = (int)((unsigned)((ga >> 32) & 0x01ffffffull) | 0x80000000u); // [56:32] | type=2
    g1[0] = (int)(1u << 16);                           // wg_mask=0, data_size=1 (2B)
    g1[1] = (int)((tensor_d0 & 0xffffu) << 16);        // tensor_dim0[15:0]
    g1[2] = (int)(((tensor_d0 >> 16) & 0xffffu) | ((tensor_d1 & 0xffffu) << 16));
    g1[3] = (int)(((tensor_d1 >> 16) & 0xffffu) | ((tile_d0 & 0xffffu) << 16));
    g1[4] = (int)(tile_d1 & 0xffffu);                  // tile_dim1; tile_dim2=0
    g1[5] = (int)row_stride_elems;                     // tensor_dim0_stride[31:0]
    g1[6] = 0;                                         // stride[47:32], dim1_stride lo
    g1[7] = 0;
    asm volatile("tensor_load_to_lds %0, %1" :: "s"(g0), "s"(g1) : "memory");
}

// ---------------- kernel: pack x[b,t,c] -> Xbf[(b*C+c), t] in bf16 ----------------
__global__ void pack_kernel(const float* __restrict__ x, bf16_t* __restrict__ Xbf) {
    int gid = blockIdx.x * blockDim.x + threadIdx.x;
    if (gid >= NROWS * TT) return;
    int n = gid >> 9, t = gid & 511;
    int b = n >> 6, c = n & 63;
    Xbf[gid] = (bf16_t)x[((size_t)b * TT + t) * CC + c];
}

// ---------------- kernel: transposed DFT basis W^T[f][k] (bf16) ----------------
// Transposed storage makes the WMMA B fragment 16 contiguous bf16 per lane
// after the TDM drops the panel into LDS (two ds_load_b128 per fragment).
__global__ void basisT_kernel(bf16_t* __restrict__ WreT, bf16_t* __restrict__ WimT,
                              int kdimLd, int cols, int denomInt,
                              int rowLimit, int colLimit, int zero_dc) {
    int gid = blockIdx.x * blockDim.x + threadIdx.x;
    if (gid >= cols * kdimLd) return;
    int f = gid / kdimLd, k = gid % kdimLd;
    float re = 0.f, im = 0.f;
    if (k < rowLimit && f < colLimit && !(zero_dc && f == 0)) {
        int ph = (int)(((long long)k * (long long)f) % (long long)denomInt);
        float ang = -6.283185307179586f * (float)ph / (float)denomInt;
        re = cosf(ang); im = sinf(ang);
    }
    WreT[gid] = (bf16_t)re; WimT[gid] = (bf16_t)im;
}

// ---------------- kernel: DFT as bf16 WMMA GEMM, TDM-staged through LDS ---------
// One wave computes a 16(M) x 32(N) tile of Y = X * W: one A fragment feeds 4
// independent WMMA chains (re/im x 2 column tiles). TDM streams the A panel
// (16 x kdim) and both transposed basis panels (32 x kdim) into LDS; every
// fragment is fed with ds_load_b128 only.
__global__ __launch_bounds__(32)
void dft_wmma_kernel(const bf16_t* __restrict__ X,
                     const bf16_t* __restrict__ WreT, const bf16_t* __restrict__ WimT,
                     float* __restrict__ Yre, float* __restrict__ Yim,
                     float* __restrict__ Amp,
                     int kdim, int ldx, int ldwT, int ldy, int namp) {
    __shared__ bf16_t sA[16 * TT];          // [16 rows][kdim]
    __shared__ bf16_t sBre[32 * TT];        // [32 cols][kdim]  (transposed basis)
    __shared__ bf16_t sBim[32 * TT];

    const int lane  = threadIdx.x;          // full wave, EXEC all ones
    const int tileN = blockIdx.x * 32;
    const int tileM = blockIdx.y * 16;

    // A panel: rows [tileM, tileM+16), cols [0, kdim)
    tdm_load_2d_b16(lds_offset_of(sA), X + (size_t)tileM * ldx,
                    (unsigned)kdim, 16u, (unsigned)kdim, 16u, (unsigned)ldx);
    // B panels (transposed): cols [tileN, tileN+32), K [0, kdim)
    tdm_load_2d_b16(lds_offset_of(sBre), WreT + (size_t)tileN * ldwT,
                    (unsigned)kdim, 32u, (unsigned)kdim, 32u, (unsigned)ldwT);
    tdm_load_2d_b16(lds_offset_of(sBim), WimT + (size_t)tileN * ldwT,
                    (unsigned)kdim, 32u, (unsigned)kdim, 32u, (unsigned)ldwT);
    __builtin_amdgcn_s_wait_tensorcnt(0);   // single-wave WG: no barrier needed

    const int mloc = lane & 15;
    const int hi   = (lane & 16) ? 1 : 0;
    const bf16_t* rowA   = sA   + (size_t)mloc * kdim;
    const bf16_t* rowB0r = sBre + (size_t)mloc * kdim;          // col tileN+mloc
    const bf16_t* rowB0i = sBim + (size_t)mloc * kdim;
    const bf16_t* rowB1r = sBre + (size_t)(mloc + 16) * kdim;   // col tileN+16+mloc
    const bf16_t* rowB1i = sBim + (size_t)(mloc + 16) * kdim;

    v8f cre[2] = {v8f{}, v8f{}};
    v8f cim[2] = {v8f{}, v8f{}};
    for (int k0 = 0; k0 < kdim; k0 += 32) {
        // A fragment: lanes 0-15 K 0-15, lanes 16-31 K 16-31 (contiguous)
        const int kA = k0 + (hi ? 16 : 0);
        union { uint4 q[2]; v16bf v; } ua;
        ua.q[0] = *(const uint4*)(rowA + kA);
        ua.q[1] = *(const uint4*)(rowA + kA + 8);

        // B fragments: lanes 0-15 hold K kB..kB+15 of their column (contiguous)
        const int kB = k0 + (hi ? 16 : 0);
        union { uint4 q[2]; v16bf v; } b0r, b0i, b1r, b1i;
        b0r.q[0] = *(const uint4*)(rowB0r + kB);
        b0r.q[1] = *(const uint4*)(rowB0r + kB + 8);
        b0i.q[0] = *(const uint4*)(rowB0i + kB);
        b0i.q[1] = *(const uint4*)(rowB0i + kB + 8);
        b1r.q[0] = *(const uint4*)(rowB1r + kB);
        b1r.q[1] = *(const uint4*)(rowB1r + kB + 8);
        b1i.q[0] = *(const uint4*)(rowB1i + kB);
        b1i.q[1] = *(const uint4*)(rowB1i + kB + 8);

        cre[0] = __builtin_amdgcn_wmma_f32_16x16x32_bf16(false, ua.v, false, b0r.v,
                                                         (short)0, cre[0], false, false);
        cim[0] = __builtin_amdgcn_wmma_f32_16x16x32_bf16(false, ua.v, false, b0i.v,
                                                         (short)0, cim[0], false, false);
        cre[1] = __builtin_amdgcn_wmma_f32_16x16x32_bf16(false, ua.v, false, b1r.v,
                                                         (short)0, cre[1], false, false);
        cim[1] = __builtin_amdgcn_wmma_f32_16x16x32_bf16(false, ua.v, false, b1i.v,
                                                         (short)0, cim[1], false, false);
    }
    // D layout: VGPR r holds M=r (lanes 0-15) / M=r+8 (lanes 16-31), N=lane&15
    const int mrow0 = tileM + (hi ? 8 : 0);
#pragma unroll
    for (int n2 = 0; n2 < 2; ++n2) {
        const int ncol = tileN + n2 * 16 + mloc;
#pragma unroll
        for (int r = 0; r < 8; ++r) {
            int mr = mrow0 + r;
            float re = cre[n2][r], im = cim[n2][r];
            if (Yre) {
                Yre[(size_t)mr * ldy + ncol] = re;
                Yim[(size_t)mr * ldy + ncol] = im;
            }
            if (Amp && ncol < namp)
                Amp[(size_t)mr * namp + ncol] = sqrtf(re * re + im * im);
        }
    }
}

// ---------------- kernel: accumulate fft-branch stats (global_atomic_add_f32) ----------------
__global__ void fft_accum_kernel(const float* __restrict__ Amp,
                                 float* __restrict__ flist, float* __restrict__ ampBF) {
    int gid = blockIdx.x * blockDim.x + threadIdx.x;
    if (gid >= NROWS * FMAIN) return;
    int n = gid / FMAIN, f = gid % FMAIN;
    float a = Amp[gid];
    atomicAdd(&flist[f], a * (1.0f / (float)NROWS));     // mean over (B,C)
    atomicAdd(&ampBF[(n >> 6) * FMAIN + f], a * (1.0f / (float)CC)); // mean over C
}

// ---------------- kernel: db4 level-1 DWT with symmetric padding ----------------
__global__ void dwt_kernel(const float* __restrict__ x,
                           bf16_t* __restrict__ cAbf, float* __restrict__ cAf) {
    int gid = blockIdx.x * blockDim.x + threadIdx.x;
    if (gid >= NROWS * KW) return;
    int n = gid / KW, j = gid % KW;
    if (j >= TC) { cAbf[gid] = (bf16_t)0.0f; return; }  // zero pad K dim for WMMA
    int b = n >> 6, c = n & 63;
    float acc = 0.f;
    // read-ahead of the strided channel gather (global_prefetch_b8)
    if (j + 8 < TC)
        __builtin_prefetch(&x[((size_t)b * TT + (2 * j + 16)) * CC + c], 0, 1);
#pragma unroll
    for (int i = 0; i < 8; ++i) {
        int p = 2 * j + 1 + i - 7;          // 'symmetric' reflection
        if (p < 0)    p = -p - 1;
        if (p >= TT)  p = 2 * TT - 1 - p;
        acc += FREV[i] * x[((size_t)b * TT + p) * CC + c];
    }
    cAbf[gid] = (bf16_t)acc;
    cAf[(size_t)n * TC + j] = acc;
}

// ---------------- kernel: wavelet per-row top-5 -> period histogram ----------------
__global__ void wav_topk_kernel(const float* __restrict__ AmpW,
                                const float* __restrict__ cAf,
                                int* __restrict__ whist, float* __restrict__ scal) {
    int n = blockIdx.x * blockDim.x + threadIdx.x;
    if (n >= NROWS) return;
    const float* amp = AmpW + (size_t)n * NW;
    int idx[KTOP];
    for (int s = 0; s < KTOP; ++s) {
        float bv = -1.f; int bi = 0;
        for (int f = 0; f < FW; ++f) {
            bool used = false;
            for (int u = 0; u < s; ++u) used |= (idx[u] == f);
            if (!used && amp[f] > bv) { bv = amp[f]; bi = f; }
        }
        idx[s] = bi;
    }
    float energy = 0.f;
    for (int j = 0; j < TC; ++j) { float v = cAf[(size_t)n * TC + j]; energy += v * v; }
    int nval = 0;
    for (int s = 0; s < KTOP; ++s) {
        int fi = idx[s];
        int p = (TC / (fi > 0 ? fi : 1)) * 2;
        bool valid = (fi > 0) && (p > 1) && (p < TT / 2);
        if (valid) {
            int pc = p < 0 ? 0 : (p > TT ? TT : p);
            atomicAdd(&whist[pc], 1);
            nval++;
        }
    }
    atomicAdd(&scal[0], energy * (float)nval);
    atomicAdd(&scal[1], (float)nval);
}

// ---------------- kernel: frequency-domain VMD (one row per block, T threads) ----
__global__ __launch_bounds__(512)
void vmd_kernel(const float* __restrict__ Yre, const float* __restrict__ Yim,
                int* __restrict__ vhist, float* __restrict__ scal) {
    __shared__ float s_red[16 * 6];
    __shared__ float s_re[TT], s_im[TT];
    __shared__ float s_max[16]; __shared__ int s_idx[16];
    __shared__ int   s_sel[2];  __shared__ float s_ene[2];

    const int n = blockIdx.x;
    const int t = threadIdx.x;              // shifted frequency index
    const int lane = t & 31, wid = t >> 5;
    const float fr = (float)t / (float)TT - 0.5f;      // freqs[t]
    const int fu = (t + TT / 2) & (TT - 1);            // unshifted index (fftshift)
    const float fhr = Yre[(size_t)n * TT + fu];
    const float fhi = Yim[(size_t)n * TT + fu];

    const float tau = 1e-7f, tau2 = 1e-14f, alpha = 2000.f;
    float omega[VK];
#pragma unroll
    for (int k = 0; k < VK; ++k) omega[k] = 0.5f * hash01((unsigned)(n * VK + k) * 0x9e3779b9u + 42u);
    float ur[VK] = {0.f, 0.f, 0.f}, ui[VK] = {0.f, 0.f, 0.f};
    float lr = 0.f, li = 0.f;

    for (int it = 0; it < VITERS; ++it) {
        float d0 = fr - omega[0]; d0 *= d0;
        float d1 = fr - omega[1]; d1 *= d1;
        float d2 = fr - omega[2]; d2 *= d2;
        float den[VK] = { d0 + tau2 + d1, d1 + tau2 + d0 + d2, d2 + tau2 + d1 };
        float sr = 0.f, si = 0.f, vals[6];
#pragma unroll
        for (int k = 0; k < VK; ++k) {
            float s = 1.f / (1.f + alpha * den[k]);
            ur[k] = (fhr - 0.5f * lr) * s;
            ui[k] = (fhi - 0.5f * li) * s;
            sr += ur[k]; si += ui[k];
            float p2 = ur[k] * ur[k] + ui[k] * ui[k];
            vals[k] = p2; vals[3 + k] = p2 * fr;
        }
#pragma unroll
        for (int j = 0; j < 6; ++j) {
            float v = wave_sum(vals[j]);
            if (lane == 0) s_red[wid * 6 + j] = v;
        }
        __syncthreads();
        float tot[6];
#pragma unroll
        for (int j = 0; j < 6; ++j) {
            float acc = 0.f;
            for (int w = 0; w < 16; ++w) acc += s_red[w * 6 + j];
            tot[j] = acc;
        }
        __syncthreads();
#pragma unroll
        for (int k = 0; k < VK; ++k) omega[k] = tot[3 + k] / (tot[k] + 1e-8f);
        lr += tau * (sr - fhr);
        li += tau * (si - fhi);
    }

    // Parseval energies: sum(imf^2) == sum|U|^2 / T
#pragma unroll
    for (int k = 0; k < VK; ++k) {
        float v = wave_sum(ur[k] * ur[k] + ui[k] * ui[k]);
        if (lane == 0) s_red[wid * 6 + k] = v;
    }
    __syncthreads();
    if (t == 0) {
        float e[VK];
        for (int k = 0; k < VK; ++k) {
            float acc = 0.f;
            for (int w = 0; w < 16; ++w) acc += s_red[w * 6 + k];
            e[k] = acc * (1.0f / (float)TT);
        }
        int i0 = 0; if (e[1] > e[i0]) i0 = 1; if (e[2] > e[i0]) i0 = 2;
        int i1 = (i0 == 0) ? 1 : 0;
        for (int k = 0; k < VK; ++k) if (k != i0 && e[k] > e[i1]) i1 = k;
        s_sel[0] = i0; s_sel[1] = i1; s_ene[0] = e[i0]; s_ene[1] = e[i1];
    }
    __syncthreads();

    // rfft amplitude of Re(ifft(U)): |U[f] + conj(U[T-f])| / 2, argmax over bins
    for (int mm = 0; mm < 2; ++mm) {
        int k = s_sel[mm];
        s_re[fu] = ur[k]; s_im[fu] = ui[k];   // store in unshifted order
        __syncthreads();
        float av = -1.f; int ai = t;
        if (t <= TT / 2) {
            int mir = (TT - t) & (TT - 1);
            float rr  = s_re[t] + s_re[mir];
            float im2 = s_im[t] - s_im[mir];
            av = 0.5f * sqrtf(rr * rr + im2 * im2);
            if (t == 0) av = 0.f;
        }
        float bv = av; int bi = ai;
        wave_argmax(bv, bi);
        if (lane == 0) { s_max[wid] = bv; s_idx[wid] = bi; }
        __syncthreads();
        if (t == 0) {
            float mv = s_max[0]; int mi = s_idx[0];
            for (int w = 1; w < 16; ++w)
                if (s_max[w] > mv || (s_max[w] == mv && s_idx[w] < mi)) { mv = s_max[w]; mi = s_idx[w]; }
            if (mi > 0) {
                int p = TT / mi;
                if (p > TT) p = TT;
                atomicAdd(&vhist[p], 1);
                atomicAdd(&scal[2], s_ene[mm]);
                atomicAdd(&scal[3], 1.0f);
            }
        }
        __syncthreads();
    }
}

// ---------------- fusion: histogram top-k with count*(T+2)+value scoring ----------------
__device__ void hist_topk5(const int* hist, int* pout, bool* anyv) {
    long long best = -1; int fb = 0; long long total = 0;
    for (int v = 0; v <= TT; ++v) {
        total += hist[v];
        long long sc = hist[v] > 0 ? (long long)hist[v] * (TT + 2) + v : -1;
        if (sc > best) { best = sc; fb = v; }
    }
    long long prev = 0x7fffffffffffffffLL;
    for (int s = 0; s < KTOP; ++s) {
        long long bs = -1; int bi = fb;
        for (int v = 0; v <= TT; ++v) {
            long long sc = hist[v] > 0 ? (long long)hist[v] * (TT + 2) + v : -1;
            if (sc >= 0 && sc < prev && sc > bs) { bs = sc; bi = v; }
        }
        pout[s] = (bs >= 0) ? bi : fb;
        if (bs >= 0) prev = bs;
    }
    *anyv = total > 0;
}

__global__ void fuse_kernel(const float* __restrict__ flist,
                            const float* __restrict__ ampBF,
                            const int* __restrict__ whist,
                            const int* __restrict__ vhist,
                            const float* __restrict__ scal,
                            float* __restrict__ out) {
    if (threadIdx.x != 0 || blockIdx.x != 0) return;

    // ---- FFT branch: top-5 frequencies of averaged spectrum (bin 0 excluded)
    int ftop[KTOP];
    bool usedf[FMAIN];
    for (int f = 0; f < FMAIN; ++f) usedf[f] = false;
    for (int s = 0; s < KTOP; ++s) {
        float bv = -1e30f; int bi = 1;
        for (int f = 1; f < FMAIN; ++f)
            if (!usedf[f] && flist[f] > bv) { bv = flist[f]; bi = f; }
        usedf[bi] = true; ftop[s] = bi;
    }
    int fftp[KTOP];
    for (int s = 0; s < KTOP; ++s) fftp[s] = TT / (ftop[s] > 0 ? ftop[s] : 1);

    // ---- wavelet & VMD branches
    int wavp[KTOP], vmdp[KTOP]; bool wav_any, vmd_any;
    hist_topk5(whist, wavp, &wav_any);
    hist_topk5(vhist, vmdp, &vmd_any);
    if (!wav_any) for (int s = 0; s < KTOP; ++s) wavp[s] = TT / 2;
    if (!vmd_any) for (int s = 0; s < KTOP; ++s) vmdp[s] = TT / 2;
    float wavw = wav_any ? (scal[0] / fmaxf(scal[1], 1.f)) / (float)KTOP : 1.0f / KTOP;
    float vmdw = vmd_any ? (scal[2] / fmaxf(scal[3], 1.f)) / (float)KTOP : 1.0f / KTOP;

    // ---- fuse candidates
    int cand[3 * KTOP]; bool cval[3 * KTOP]; int cc[3 * KTOP]; long long sc[3 * KTOP];
    for (int s = 0; s < KTOP; ++s) {
        cand[s] = fftp[s]; cand[KTOP + s] = wavp[s]; cand[2 * KTOP + s] = vmdp[s];
    }
    bool any = false;
    for (int i = 0; i < 3 * KTOP; ++i) {
        cval[i] = (cand[i] > 1) && (cand[i] < TT / 2);
        any |= cval[i];
        int v = cand[i]; if (v < 0) v = 0; if (v > TT) v = TT;
        cc[i] = v;
    }
    long long best = -1; int fbval = 0;
    for (int i = 0; i < 3 * KTOP; ++i) {
        int cnt = 0;
        for (int j = 0; j < 3 * KTOP; ++j) if (cval[j] && cc[j] == cc[i]) cnt++;
        sc[i] = cval[i] ? (long long)cnt * (TT + 2) + cc[i] : -1;
        if (sc[i] > best) { best = sc[i]; fbval = cc[i]; }
    }
    int fusedp[KTOP];
    long long prev = 0x7fffffffffffffffLL;
    for (int s = 0; s < KTOP; ++s) {
        long long bs = -1; int bi = -1;
        for (int i = 0; i < 3 * KTOP; ++i)
            if (sc[i] >= 0 && sc[i] < prev && sc[i] > bs) { bs = sc[i]; bi = i; }
        fusedp[s] = (bs >= 0) ? cc[bi] : fbval;
        if (bs >= 0) prev = bs;
    }
    if (!any) for (int s = 0; s < KTOP; ++s) fusedp[s] = fftp[s];

    // ---- per-method presence and weight fusion
    float pf[KTOP], pw[KTOP], pv[KTOP]; int idxf[KTOP];
    for (int k = 0; k < KTOP; ++k) {
        pf[k] = 0.f; pw[k] = 0.f; pv[k] = 0.f; idxf[k] = 0;
        for (int j = 0; j < KTOP; ++j) if (fusedp[k] == fftp[j]) { pf[k] = 1.f; idxf[k] = j; break; }
        for (int j = 0; j < KTOP; ++j) if (fusedp[k] == wavp[j]) { pw[k] = 1.f; break; }
        for (int j = 0; j < KTOP; ++j) if (fusedp[k] == vmdp[j]) { pv[k] = 1.f; break; }
    }
    for (int k = 0; k < KTOP; ++k) out[k] = (float)fusedp[k];
    for (int b = 0; b < BN; ++b) {
        float fbw = 0.f;
        for (int k = 0; k < KTOP; ++k) fbw += ampBF[b * FMAIN + ftop[k]];
        fbw *= 1.0f / (float)KTOP;
        for (int k = 0; k < KTOP; ++k) {
            float cnt = pf[k] + pw[k] + pv[k];
            float contrib = ampBF[b * FMAIN + ftop[idxf[k]]] * pf[k] + wavw * pw[k] + vmdw * pv[k];
            float w = (cnt > 0.f) ? contrib / fmaxf(cnt, 1.f) : fbw;
            if (!any) w = ampBF[b * FMAIN + ftop[k]];
            out[KTOP + b * KTOP + k] = w;
        }
    }
}

// ---------------- host launcher ----------------
extern "C" void kernel_launch(void* const* d_in, const int* in_sizes, int n_in,
                              void* d_out, int out_size, void* d_ws, size_t ws_size,
                              hipStream_t stream) {
    (void)in_sizes; (void)n_in; (void)out_size; (void)ws_size;
    const float* x = (const float*)d_in[0];
    float* out = (float*)d_out;
    char* ws = (char*)d_ws;

    size_t off = 0;
    auto take = [&](size_t bytes) -> size_t {
        size_t r = off; off += (bytes + 255) & ~(size_t)255; return r;
    };
    size_t oXbf  = take((size_t)NROWS * TT * 2);
    size_t oWre  = take((size_t)TT * TT * 2);      // transposed: [f][k]
    size_t oWim  = take((size_t)TT * TT * 2);
    size_t oYre  = take((size_t)NROWS * TT * 4);
    size_t oYim  = take((size_t)NROWS * TT * 4);
    size_t oAmpM = take((size_t)NROWS * FMAIN * 4);
    size_t oCAbf = take((size_t)NROWS * KW * 2);
    size_t oCAf  = take((size_t)NROWS * TC * 4);
    size_t oW2re = take((size_t)NW * KW * 2);      // transposed: [f][k]
    size_t oW2im = take((size_t)NW * KW * 2);
    size_t oAmpW = take((size_t)NROWS * NW * 4);
    // contiguous zeroed accumulators (4-byte words)
    const size_t ACC_FLIST = 0;
    const size_t ACC_AMPBF = ACC_FLIST + FMAIN;
    const size_t ACC_WHIST = ACC_AMPBF + (size_t)BN * FMAIN;
    const size_t ACC_VHIST = ACC_WHIST + (TT + 1);
    const size_t ACC_SCAL  = ACC_VHIST + (TT + 1);
    const size_t ACC_WORDS = ACC_SCAL + 8;
    size_t oACC = take(ACC_WORDS * 4);

    bf16_t* Xbf  = (bf16_t*)(ws + oXbf);
    bf16_t* WreT = (bf16_t*)(ws + oWre);
    bf16_t* WimT = (bf16_t*)(ws + oWim);
    float*  Yre  = (float*)(ws + oYre);
    float*  Yim  = (float*)(ws + oYim);
    float*  AmpM = (float*)(ws + oAmpM);
    bf16_t* cAbf = (bf16_t*)(ws + oCAbf);
    float*  cAf  = (float*)(ws + oCAf);
    bf16_t* W2reT = (bf16_t*)(ws + oW2re);
    bf16_t* W2imT = (bf16_t*)(ws + oW2im);
    float*  AmpW = (float*)(ws + oAmpW);
    float*  accF = (float*)(ws + oACC);
    float*  flist = accF + ACC_FLIST;
    float*  ampBF = accF + ACC_AMPBF;
    int*    whist = (int*)(accF + ACC_WHIST);
    int*    vhist = (int*)(accF + ACC_VHIST);
    float*  scal  = accF + ACC_SCAL;

    hipMemsetAsync(ws + oACC, 0, ACC_WORDS * 4, stream);   // graph-capturable

    // build transposed DFT bases
    {
        int n1 = TT * TT;
        basisT_kernel<<<(n1 + 255) / 256, 256, 0, stream>>>(WreT, WimT, TT, TT, TT, TT, TT, 0);
        int n2 = NW * KW;
        basisT_kernel<<<(n2 + 255) / 256, 256, 0, stream>>>(W2reT, W2imT, KW, NW, TC, TC, FW, 1);
    }
    // pack and DWT
    pack_kernel<<<(NROWS * TT + 255) / 256, 256, 0, stream>>>(x, Xbf);
    dwt_kernel<<<(NROWS * KW + 255) / 256, 256, 0, stream>>>(x, cAbf, cAf);

    // main DFT (full complex spectrum for VMD + rfft amplitudes), WMMA + TDM
    dft_wmma_kernel<<<dim3(TT / 32, NROWS / 16), 32, 0, stream>>>(
        Xbf, WreT, WimT, Yre, Yim, AmpM, TT, TT, TT, TT, FMAIN);
    // wavelet DFT (amplitudes only), WMMA + TDM
    dft_wmma_kernel<<<dim3(NW / 32, NROWS / 16), 32, 0, stream>>>(
        cAbf, W2reT, W2imT, nullptr, nullptr, AmpW, KW, KW, KW, 0, NW);

    // branch statistics
    fft_accum_kernel<<<(NROWS * FMAIN + 255) / 256, 256, 0, stream>>>(AmpM, flist, ampBF);
    wav_topk_kernel<<<(NROWS + 255) / 256, 256, 0, stream>>>(AmpW, cAf, whist, scal);
    vmd_kernel<<<NROWS, 512, 0, stream>>>(Yre, Yim, vhist, scal);

    // final fusion -> d_out: [5] periods (as float), then [16 x 5] weights
    fuse_kernel<<<1, 32, 0, stream>>>(flist, ampBF, whist, vhist, scal, out);
}
